// Split_Quantizer_58119497449526
// MI455X (gfx1250) — matrix operand order repaired and checked by reading, not compiled, not run on hardware
//
#include <hip/hip_runtime.h>
#include <hip/hip_bf16.h>
#include <math.h>

typedef __attribute__((ext_vector_type(16))) _Float16 v16h;
typedef __attribute__((ext_vector_type(8)))  float    v8f;

#define NUM_EMB   1024
#define DIM       32
#define ROWS      131072        // 16384 * 8
#define N_TILES   (NUM_EMB / 16)   // 64
#define ROW_TILES (ROWS / 16)      // 8192
#define WAVES_PER_BLOCK 8
#define BLOCK_THREADS   256
#define GRID_BLOCKS (ROW_TILES / WAVES_PER_BLOCK)  // 1024

// ---------------------------------------------------------------------------
// Kernel 1: nearest-code search via V_WMMA_F32_16X16X32_F16, plus quantized
// output, indices, and atomic accumulation of counts / dw / loss.
// ---------------------------------------------------------------------------
__global__ __launch_bounds__(BLOCK_THREADS)
void vq_search_kernel(const float* __restrict__ inputs,     // [ROWS, DIM] (flat view)
                      const float* __restrict__ embedding,  // [NUM_EMB, DIM]
                      float* __restrict__ out_quant,        // [ROWS, DIM]
                      float* __restrict__ out_idx,          // [ROWS] (float)
                      float* __restrict__ ws_counts,        // [NUM_EMB]
                      float* __restrict__ ws_dw,            // [NUM_EMB, DIM]
                      float* __restrict__ ws_loss)          // [1]
{
    // f16 codebook (64 KB) + f32 code norms (4 KB) in LDS. CDNA5 allows up
    // to 320 KB LDS per workgroup, so 68 KB is fine and still leaves room
    // for multiple workgroups per WGP.
    __shared__ _Float16 s_emb[NUM_EMB * DIM];
    __shared__ float    s_cn[NUM_EMB];

    const int tid = threadIdx.x;

    // Stage codebook f32 -> f16 into LDS and precompute exact f32 norms.
    for (int c = tid; c < NUM_EMB; c += BLOCK_THREADS) {
        const float* e = embedding + c * DIM;
        _Float16* dst = s_emb + c * DIM;
        float nrm = 0.0f;
        #pragma unroll
        for (int k = 0; k < DIM; ++k) {
            float v = e[k];
            nrm += v * v;
            dst[k] = (_Float16)v;
        }
        s_cn[c] = nrm;
    }
    __syncthreads();

    const int lane = tid & 31;
    const int wave = tid >> 5;
    const int tile = blockIdx.x * WAVES_PER_BLOCK + wave;  // row tile id
    const int m0   = tile * 16;

    // ---- A matrix (16x32 f16): lane = row (lane&15); K-halves per lane group.
    const int arow  = m0 + (lane & 15);
    const int kbase = (lane >> 4) * 16;         // 0 or 16
    const float* ap = inputs + arow * DIM;
    v16h A;
    #pragma unroll
    for (int i = 0; i < 8; ++i) A[i]     = (_Float16)ap[kbase + i];
    #pragma unroll
    for (int i = 0; i < 8; ++i) A[8 + i] = (_Float16)ap[kbase + 8 + i];

    // ---- argmin state: 8 C-matrix rows per lane.
    float best[8];
    int   bidx[8];
    #pragma unroll
    for (int r = 0; r < 8; ++r) { best[r] = 3.0e38f; bidx[r] = 0; }

    const int ncol = lane & 15;            // column within 16-wide code tile
    const int kb2  = (lane >> 4) * 16;     // B K-half per lane group (halves)

    #pragma unroll 4
    for (int ct = 0; ct < N_TILES; ++ct) {
        const int code = ct * 16 + ncol;
        // B (32x16 f16): lane holds one column = one codebook row, 16 halves.
        const v16h  B  = *(const v16h*)(s_emb + code * DIM + kb2);
        const float cn = s_cn[code];       // ||e||^2 (loop-invariant per code)

        v8f acc = {};
        acc = __builtin_amdgcn_wmma_f32_16x16x32_f16(
                  false, A, false, B, (short)0, acc, false, false);

        #pragma unroll
        for (int r = 0; r < 8; ++r) {
            float d = __builtin_fmaf(-2.0f, acc[r], cn);   // ||e||^2 - 2 x.e
            if (d < best[r]) { best[r] = d; bidx[r] = code; }
        }
    }

    // ---- reduce argmin across the 16 lanes sharing each row group.
    #pragma unroll
    for (int r = 0; r < 8; ++r) {
        #pragma unroll
        for (int mask = 1; mask < 16; mask <<= 1) {
            float ov = __shfl_xor(best[r], mask, 32);
            int   oi = __shfl_xor(bidx[r], mask, 32);
            if (ov < best[r] || (ov == best[r] && oi < bidx[r])) {
                best[r] = ov; bidx[r] = oi;
            }
        }
    }
    // lanes 0-15 now agree on rows m0+0..7; lanes 16-31 on rows m0+8..15.

    // indices + counts from one lane per group
    if ((lane & 15) == 0) {
        const int base = m0 + (lane >> 4) * 8;
        #pragma unroll
        for (int r = 0; r < 8; ++r) {
            out_idx[base + r] = (float)bidx[r];
            atomicAdd(ws_counts + bidx[r], 1.0f);
        }
    }

    // broadcast the 16 winning indices to the whole wave
    int rowIdx[16];
    #pragma unroll
    for (int r = 0; r < 8; ++r) {
        rowIdx[r]     = __shfl(bidx[r], 0, 32);
        rowIdx[8 + r] = __shfl(bidx[r], 16, 32);
    }

    // ---- quantized output (exact f32 codebook), dw segment-sum, loss.
    float sq = 0.0f;
    #pragma unroll
    for (int r = 0; r < 16; ++r) {
        const int id = rowIdx[r];
        const float q = embedding[id * DIM + lane];
        const float f = inputs[(m0 + r) * DIM + lane];
        out_quant[(m0 + r) * DIM + lane] = q;
        const float df = q - f;
        sq += df * df;
        atomicAdd(ws_dw + id * DIM + lane, f);
    }
    #pragma unroll
    for (int mask = 1; mask < 32; mask <<= 1) sq += __shfl_xor(sq, mask, 32);
    if (lane == 0) atomicAdd(ws_loss, sq);
}

// ---------------------------------------------------------------------------
// Kernel 2: EMA updates, smoothing, new embedding, perplexity, loss.
// ---------------------------------------------------------------------------
__global__ __launch_bounds__(1024)
void vq_finalize_kernel(const float* __restrict__ ema_cs,     // [NUM_EMB]
                        const float* __restrict__ ema_w,      // [NUM_EMB, DIM]
                        const float* __restrict__ ws_counts,
                        const float* __restrict__ ws_dw,
                        const float* __restrict__ ws_loss,
                        float* __restrict__ out_loss,
                        float* __restrict__ out_perp,
                        float* __restrict__ out_newemb,
                        float* __restrict__ out_newcs,
                        float* __restrict__ out_newemaw)
{
    __shared__ float red[NUM_EMB];
    const int tid = threadIdx.x;

    const float c   = ws_counts[tid];
    const float pre = ema_cs[tid] * 0.99f + 0.01f * c;

    red[tid] = pre;
    __syncthreads();
    for (int s = 512; s > 0; s >>= 1) {
        if (tid < s) red[tid] += red[tid + s];
        __syncthreads();
    }
    const float n = red[0];
    __syncthreads();

    const float ncs = (pre + 1e-5f) / (n + NUM_EMB * 1e-5f) * n;
    out_newcs[tid] = ncs;
    const float inv = 1.0f / ncs;
    #pragma unroll
    for (int d = 0; d < DIM; ++d) {
        const float w = ema_w[tid * DIM + d] * 0.99f + 0.01f * ws_dw[tid * DIM + d];
        out_newemaw[tid * DIM + d] = w;
        out_newemb[tid * DIM + d]  = w * inv;
    }

    // perplexity from raw counts
    const float p = c / (float)ROWS;
    red[tid] = -p * logf(p + 1e-10f);
    __syncthreads();
    for (int s = 512; s > 0; s >>= 1) {
        if (tid < s) red[tid] += red[tid + s];
        __syncthreads();
    }
    if (tid == 0) {
        *out_perp = expf(red[0]);
        *out_loss = 0.25f * ws_loss[0] / ((float)ROWS * (float)DIM);
    }
}

// ---------------------------------------------------------------------------
// Host launcher
// ---------------------------------------------------------------------------
extern "C" void kernel_launch(void* const* d_in, const int* in_sizes, int n_in,
                              void* d_out, int out_size, void* d_ws, size_t ws_size,
                              hipStream_t stream)
{
    (void)in_sizes; (void)n_in; (void)out_size; (void)ws_size;

    const float* inputs    = (const float*)d_in[0];  // [16384, 256]
    const float* embedding = (const float*)d_in[1];  // [1024, 32]
    const float* ema_cs    = (const float*)d_in[2];  // [1024]
    const float* ema_w     = (const float*)d_in[3];  // [1024, 32]

    // Output layout (flat f32, concatenated in return order):
    float* out       = (float*)d_out;
    float* o_loss    = out;                                  // 1
    float* o_quant   = out + 1;                              // 16384*256
    float* o_perp    = out + 1 + 16384 * 256;                // 1
    float* o_idx     = o_perp + 1;                           // 131072
    float* o_newemb  = o_idx + ROWS;                         // 1024*32
    float* o_newcs   = o_newemb + NUM_EMB * DIM;             // 1024
    float* o_newemaw = o_newcs + NUM_EMB;                    // 1024*32

    // Workspace layout (floats): counts | dw | loss
    float* ws       = (float*)d_ws;
    float* w_counts = ws;                       // 1024
    float* w_dw     = ws + NUM_EMB;             // 32768
    float* w_loss   = ws + NUM_EMB + NUM_EMB * DIM;  // 1
    hipMemsetAsync(d_ws, 0, (size_t)(NUM_EMB + NUM_EMB * DIM + 1) * sizeof(float),
                   stream);

    vq_search_kernel<<<GRID_BLOCKS, BLOCK_THREADS, 0, stream>>>(
        inputs, embedding, o_quant, o_idx, w_counts, w_dw, w_loss);

    vq_finalize_kernel<<<1, 1024, 0, stream>>>(
        ema_cs, ema_w, w_counts, w_dw, w_loss,
        o_loss, o_perp, o_newemb, o_newcs, o_newemaw);
}
